// Cosine_Affinity_8177617731969
// MI455X (gfx1250) — compile-verified, matrix operand order checked by reading, not run on hardware
//
#include <hip/hip_runtime.h>
#include <hip/hip_bf16.h>

// Problem constants (fixed by the reference: B=8, N=M=2048, d=128)
constexpr int Bc = 8;
constexpr int Nc = 2048;
constexpr int Mc = 2048;
constexpr int Dc = 128;

typedef _Float16 v4h  __attribute__((ext_vector_type(4)));
typedef _Float16 v8h  __attribute__((ext_vector_type(8)));
typedef _Float16 v16h __attribute__((ext_vector_type(16)));
typedef float    v8f  __attribute__((ext_vector_type(8)));

// ---------------------------------------------------------------------------
// Kernel 1: gated row-normalize, fp32 -> fp16.
// One wave32 per row of d=128 (4 floats per lane). Wave-reduce sum of squares,
// scale by rsqrt(max(ss, eps)), emit packed _Float16.
// ---------------------------------------------------------------------------
__global__ __launch_bounds__(256) void gate_normalize_f16(
    const float* __restrict__ X, const float* __restrict__ G,
    _Float16* __restrict__ out, int rows)
{
    const int row  = blockIdx.x * 8 + (threadIdx.x >> 5);
    const int lane = threadIdx.x & 31;
    if (row >= rows) return;

    const float4 x = *((const float4*)(X + (size_t)row * Dc) + lane);
    const float4 g = *((const float4*)(G + (size_t)row * Dc) + lane);

    const float g0 = x.x * g.x;
    const float g1 = x.y * g.y;
    const float g2 = x.z * g.z;
    const float g3 = x.w * g.w;

    float ss = g0 * g0 + g1 * g1 + g2 * g2 + g3 * g3;
    #pragma unroll
    for (int off = 16; off > 0; off >>= 1)
        ss += __shfl_xor(ss, off, 32);

    const float scale = rsqrtf(fmaxf(ss, 1e-6f));

    v4h o;
    o.x = (_Float16)(g0 * scale);
    o.y = (_Float16)(g1 * scale);
    o.z = (_Float16)(g2 * scale);
    o.w = (_Float16)(g3 * scale);
    *((v4h*)(out + (size_t)row * Dc) + lane) = o;
}

// ---------------------------------------------------------------------------
// Kernel 2: batched GEMM via v_wmma_f32_16x16x32_f16.
// out[b,n,m] = sum_k Xn[b,n,k] * Yn[b,m,k]
// Block = 256 threads (8 waves), 128(n) x 128(m) output tile.
// Waves arranged 4(n) x 2(m); each wave owns 2x4 tiles of 16x16.
//
// Fragment gather follows the CDNA5 16-bit A-matrix layout (ISA 7.12.2):
//   lane L (<16):  row = base+L, halfs K=[k0, k0+8) in VGPR0-3, [k0+16, k0+24) in VGPR4-7
//   lane L (>=16): row = base+L-16, halfs K=[k0+8, k0+16) and [k0+24, k0+32)
// Since Yn is row-major and B-columns == Yn-rows, B uses the same gather.
// ---------------------------------------------------------------------------
__global__ __launch_bounds__(256) void cosine_gemm_wmma(
    const _Float16* __restrict__ Xn, const _Float16* __restrict__ Yn,
    float* __restrict__ out)
{
    const int b     = blockIdx.z;
    const int mBase = blockIdx.x * 128;          // columns (Y rows)
    const int nBase = blockIdx.y * 128;          // rows    (X rows)
    const int wave  = threadIdx.x >> 5;          // 0..7
    const int lane  = threadIdx.x & 31;
    const int waveN = wave >> 1;                 // 0..3 -> 32 rows each
    const int waveM = wave & 1;                  // 0..1 -> 64 cols each
    const int laneLo  = lane & 15;               // row/col within 16-tile
    const int laneKhi = (lane >> 4) * 8;         // K sub-offset: 0 or 8

    const _Float16* Xb = Xn + (size_t)b * Nc * Dc;
    const _Float16* Yb = Yn + (size_t)b * Mc * Dc;

    const int n0 = nBase + waveN * 32;
    const int m0 = mBase + waveM * 64;

    v8f acc[2][4];
    #pragma unroll
    for (int i = 0; i < 2; ++i)
        #pragma unroll
        for (int j = 0; j < 4; ++j)
            acc[i][j] = (v8f)0.0f;

    #pragma unroll
    for (int k0 = 0; k0 < Dc; k0 += 32) {
        const int k = k0 + laneKhi;

        v16h afr[2];
        #pragma unroll
        for (int i = 0; i < 2; ++i) {
            const _Float16* p = Xb + (size_t)(n0 + i * 16 + laneLo) * Dc + k;
            v8h lo = *(const v8h*)p;
            v8h hi = *(const v8h*)(p + 16);
            afr[i] = __builtin_shufflevector(lo, hi,
                0, 1, 2, 3, 4, 5, 6, 7, 8, 9, 10, 11, 12, 13, 14, 15);
        }

        v16h bfr[4];
        #pragma unroll
        for (int j = 0; j < 4; ++j) {
            const _Float16* p = Yb + (size_t)(m0 + j * 16 + laneLo) * Dc + k;
            v8h lo = *(const v8h*)p;
            v8h hi = *(const v8h*)(p + 16);
            bfr[j] = __builtin_shufflevector(lo, hi,
                0, 1, 2, 3, 4, 5, 6, 7, 8, 9, 10, 11, 12, 13, 14, 15);
        }

        #pragma unroll
        for (int i = 0; i < 2; ++i)
            #pragma unroll
            for (int j = 0; j < 4; ++j)
                acc[i][j] = __builtin_amdgcn_wmma_f32_16x16x32_f16(
                    /*neg_a=*/false, afr[i], /*neg_b=*/false, bfr[j],
                    /*c_mod=*/(short)0, acc[i][j],
                    /*reuse_a=*/false, /*reuse_b=*/false);
    }

    // Store per C/D layout: VGPR r -> row r (lanes 0-15) / row r+8 (lanes 16-31),
    // lane&15 -> column within tile.
    const int rowHalf = (lane >> 4) * 8;
    #pragma unroll
    for (int i = 0; i < 2; ++i) {
        #pragma unroll
        for (int j = 0; j < 4; ++j) {
            const int m = m0 + j * 16 + laneLo;
            #pragma unroll
            for (int r = 0; r < 8; ++r) {
                const int n = n0 + i * 16 + rowHalf + r;
                out[((size_t)b * Nc + n) * Mc + m] = acc[i][j][r];
            }
        }
    }
}

extern "C" void kernel_launch(void* const* d_in, const int* in_sizes, int n_in,
                              void* d_out, int out_size, void* d_ws, size_t ws_size,
                              hipStream_t stream) {
    const float* X  = (const float*)d_in[0];
    const float* Y  = (const float*)d_in[1];
    const float* A1 = (const float*)d_in[2];
    const float* A2 = (const float*)d_in[3];
    float* out = (float*)d_out;

    _Float16* Xn = (_Float16*)d_ws;                         // B*N*D halfs (4 MiB)
    _Float16* Yn = Xn + (size_t)Bc * Nc * Dc;               // B*M*D halfs (4 MiB)

    // Phase 1: gate + normalize, fp32 -> fp16 (8 rows per 256-thread block)
    gate_normalize_f16<<<(Bc * Nc) / 8, 256, 0, stream>>>(X, A1, Xn, Bc * Nc);
    gate_normalize_f16<<<(Bc * Mc) / 8, 256, 0, stream>>>(Y, A2, Yn, Bc * Mc);

    // Phase 2: batched cosine GEMM with WMMA
    dim3 grid(Mc / 128, Nc / 128, Bc);
    cosine_gemm_wmma<<<grid, 256, 0, stream>>>(Xn, Yn, out);
}
